// CrossWinAttention_8684423872893
// MI455X (gfx1250) — matrix operand order repaired and checked by reading, not compiled, not run on hardware
//
#include <hip/hip_runtime.h>
#include <hip/hip_bf16.h>

// ---------------- problem constants ----------------
#define B_    2
#define CAM_  6
#define L_    16       // X*Y
#define S_    384      // n*w1*w2
#define DIM_  512
#define H_    16
#define DH_   32
#define ROWS_ (B_*L_*S_)   // 12288

typedef __attribute__((ext_vector_type(16))) __bf16 v16bf;
typedef __attribute__((ext_vector_type(8)))  float  v8f;

union FragU { v16bf v; uint4 q[2]; };

// ---- CDNA5 async global->LDS copy via inline asm (ISA §15.18 opcode 98) ----
#define USE_ASYNC_LDS 1

__device__ __forceinline__ void async_copy16(const void* g, void* l) {
#if USE_ASYNC_LDS
  unsigned lds_addr = (unsigned)(uintptr_t)l;
  asm volatile("global_load_async_to_lds_b128 %0, %1, off"
               :: "v"(lds_addr), "v"(g) : "memory");
#else
  *(uint4*)l = *(const uint4*)g;
#endif
}

__device__ __forceinline__ void async_wait_barrier() {
#if USE_ASYNC_LDS
  asm volatile("s_wait_asynccnt 0" ::: "memory");
#endif
  __syncthreads();
}

__device__ __forceinline__ unsigned short f2bf(float f) {
  unsigned int u = __float_as_uint(f);
  u += 0x7FFFu + ((u >> 16) & 1u);     // round-to-nearest-even
  return (unsigned short)(u >> 16);
}
__device__ __forceinline__ float bf2f(unsigned short u) {
  return __uint_as_float((unsigned)u << 16);
}

__device__ __forceinline__ v8f wmma_bf16(v16bf a, v16bf b, v8f c) {
  return __builtin_amdgcn_wmma_f32_16x16x32_bf16(
      false, a, false, b, (short)0, c, false, false);
}

// A fragment: 16x32 bf16, row-major. Lane l holds row (l&15); kb=(l<16)?0:8.
template <typename T>
__device__ __forceinline__ v16bf load_fragA(const T* base, int ld, int lane) {
  const T* p = base + (lane & 15) * ld + ((lane & 16) >> 1);
  FragU f;
  f.q[0] = *(const uint4*)(p);
  f.q[1] = *(const uint4*)(p + 16);
  return f.v;
}

// B fragment: 32x16 bf16 from Bt = B^T (N x K row-major).
template <typename T>
__device__ __forceinline__ v16bf load_fragB(const T* base, int ld, int lane) {
  const T* p = base + (lane & 15) * ld + ((lane & 16) >> 1);
  FragU f;
  f.q[0] = *(const uint4*)(p);
  f.q[1] = *(const uint4*)(p + 16);
  return f.v;
}

// ---------------- kernel 1: weight convert + transpose (fp32 KxN -> bf16 NxK)
__global__ void wt_convert_kernel(const float* __restrict__ W,
                                  unsigned short* __restrict__ Wt) {
  int idx = blockIdx.x * 256 + threadIdx.x;   // over 512*512
  int k = idx >> 9, n = idx & 511;
  Wt[(size_t)n * 512 + k] = f2bf(W[idx]);
}

// ---------------- kernel 2: gather-permute + LayerNorm + bf16 ----------------
__global__ void ln_gather_kernel(const float* __restrict__ src,
                                 const float* __restrict__ g,
                                 const float* __restrict__ bta,
                                 unsigned short* __restrict__ dst) {
  int row = blockIdx.x;                // [0, ROWS_)
  int tid = threadIdx.x;               // 256
  int bb  = row / (L_ * S_);
  int rem = row % (L_ * S_);
  int l = rem / S_, s = rem % S_;
  int x = l >> 2, y = l & 3;
  int cam = s / 64, p = s & 63, wi = p >> 3, wj = p & 7;
  const float* sp = src +
      ((((((size_t)bb * CAM_ + cam) * 4 + x) * 4 + y) * 8 + wi) * 8 + wj) * DIM_;
  float v0 = sp[tid], v1 = sp[tid + 256];

  __shared__ float red[256];
  red[tid] = v0 + v1;  __syncthreads();
  for (int o = 128; o; o >>= 1) { if (tid < o) red[tid] += red[tid + o]; __syncthreads(); }
  float mu = red[0] * (1.f / 512.f);
  __syncthreads();
  float d0 = v0 - mu, d1 = v1 - mu;
  red[tid] = d0 * d0 + d1 * d1;  __syncthreads();
  for (int o = 128; o; o >>= 1) { if (tid < o) red[tid] += red[tid + o]; __syncthreads(); }
  float rstd = rsqrtf(red[0] * (1.f / 512.f) + 1e-5f);

  unsigned short* dr = dst + (size_t)row * DIM_;
  dr[tid]       = f2bf(d0 * rstd * g[tid]       + bta[tid]);
  dr[tid + 256] = f2bf(d1 * rstd * g[tid + 256] + bta[tid + 256]);
}

// ---------------- kernel 3: bf16 WMMA GEMM (M x 512) * (512 x 512) ----------
// Wave tile 16x128 (8 accumulators = 64 VGPRs -> no spills, 8x A reuse).
__global__ __launch_bounds__(128)
void gemm_bf16_kernel(const unsigned short* __restrict__ A,
                      const unsigned short* __restrict__ Bt,
                      const float* __restrict__ bias,
                      void* __restrict__ out, int mode) {
  int lane = threadIdx.x & 31;
  int wave = threadIdx.x >> 5;                 // 4 waves
  int m0 = blockIdx.x * 64 + wave * 16;
  int n0 = blockIdx.y * 128;
  v8f acc[8] = {};
  const unsigned short* Ab = A + (size_t)m0 * 512;
  for (int kk = 0; kk < 512; kk += 32) {
    __builtin_prefetch(Ab + kk + 128, 0, 1);               // global_prefetch_b8
    v16bf af = load_fragA(Ab + kk, 512, lane);
#pragma unroll
    for (int nt = 0; nt < 8; ++nt) {
      v16bf bf = load_fragB(Bt + (size_t)(n0 + nt * 16) * 512 + kk, 512, lane);
      acc[nt] = wmma_bf16(af, bf, acc[nt]);
    }
  }
  int ncol = lane & 15;
  int mb   = (lane >> 4) * 8;
#pragma unroll
  for (int nt = 0; nt < 8; ++nt) {
#pragma unroll
    for (int r = 0; r < 8; ++r) {
      int m = m0 + mb + r;
      int n = n0 + nt * 16 + ncol;
      float val = acc[nt][r] + bias[n];
      if (mode == 0) {
        ((float*)out)[(size_t)m * 512 + n] = val;
      } else {
        int bb = m / (L_ * S_);
        int rem = m % (L_ * S_);
        int l = rem / S_, s = rem % S_;
        int h = n >> 5, e = n & 31;
        size_t idx = (mode == 1)
            ? ((((size_t)(bb * H_ + h) * L_ + l) * S_ + s) * DH_ + e)
            : ((((size_t)(bb * H_ + h) * L_ + l) * DH_ + e) * S_ + s);
        ((unsigned short*)out)[idx] = f2bf(val);
      }
    }
  }
}

// ---------------- kernel 4: fused attention per (b,h,l) ---------------------
// K/V staged once per block into LDS (async copy). Scores computed in 3 chunks
// of 8 WMMA tiles (low register pressure), raw scaled scores parked in LDS,
// exp applied in a second LDS pass, 1/rowsum folded into the PV epilogue.
__global__ __launch_bounds__(128)
void attn_kernel(const unsigned short* __restrict__ qh,
                 const unsigned short* __restrict__ kh,
                 const unsigned short* __restrict__ vt,
                 const float* __restrict__ lscale,
                 float* __restrict__ aout) {
  int bid = blockIdx.x;            // b*H*L = 512
  int bb = bid / (H_ * L_);
  int h  = (bid / L_) % H_;
  int l  = bid % L_;
  int lane = threadIdx.x & 31;
  int wave = threadIdx.x >> 5;     // 4 waves
  size_t blk = (size_t)(bb * H_ + h) * L_ + l;
  const unsigned short* Q = qh + blk * (S_ * DH_);
  const unsigned short* K = kh + blk * (S_ * DH_);
  const unsigned short* V = vt + blk * (DH_ * S_);
  float* O = aout + ((size_t)(bb * L_ + l) * S_) * DIM_ + h * DH_;
  // Faithful to reference: scale indexed by the L axis (H==L broadcast quirk)
  float sc = __expf(lscale[l]) * 0.17677669529663687f;   // 1/sqrt(32)

  __shared__ __align__(16) unsigned short Ks[S_ * DH_];   // 24 KB
  __shared__ __align__(16) unsigned short Vs[DH_ * S_];   // 24 KB
  __shared__ __align__(16) unsigned short P[4][16][S_];   // 48 KB (per-wave)
  __shared__ float rowmax_s[4][16];
  __shared__ float rowsum_s[4][16];

  // ---- stage K and V into LDS (1536 x 16B chunks each, 128 threads) ----
  for (int i = threadIdx.x; i < (S_ * DH_) / 8; i += 128) {
    async_copy16(K + (size_t)i * 8, Ks + (size_t)i * 8);
    async_copy16(V + (size_t)i * 8, Vs + (size_t)i * 8);
  }
  async_wait_barrier();

  int ncol = lane & 15;
  int mb   = (lane >> 4) * 8;

  for (int rt = wave; rt < S_ / 16; rt += 4) {           // 6 row-tiles / wave
    int m0 = rt * 16;
    v16bf af = load_fragA(Q + (size_t)m0 * DH_, DH_, lane);
    float rmax[8];
#pragma unroll
    for (int r = 0; r < 8; ++r) rmax[r] = -3.0e38f;

    // ---- pass A: scores in 3 chunks of 8 tiles; raw scaled scores -> LDS ----
    for (int ch = 0; ch < 3; ++ch) {
      v8f acc[8] = {};
#pragma unroll
      for (int nt = 0; nt < 8; ++nt) {
        v16bf bf = load_fragB(Ks + (size_t)((ch * 8 + nt) * 16) * DH_, DH_, lane);
        acc[nt] = wmma_bf16(af, bf, acc[nt]);
      }
#pragma unroll
      for (int nt = 0; nt < 8; ++nt)
#pragma unroll
        for (int r = 0; r < 8; ++r) {
          float x = acc[nt][r] * sc;
          rmax[r] = fmaxf(rmax[r], x);
          P[wave][mb + r][(ch * 8 + nt) * 16 + ncol] = f2bf(x);
        }
    }
    // per-row max across the 16-lane halves -> LDS
#pragma unroll
    for (int r = 0; r < 8; ++r) {
      rmax[r] = fmaxf(rmax[r], __shfl_xor(rmax[r], 8, 16));
      rmax[r] = fmaxf(rmax[r], __shfl_xor(rmax[r], 4, 16));
      rmax[r] = fmaxf(rmax[r], __shfl_xor(rmax[r], 2, 16));
      rmax[r] = fmaxf(rmax[r], __shfl_xor(rmax[r], 1, 16));
    }
    if (ncol == 0) {
#pragma unroll
      for (int r = 0; r < 8; ++r) rowmax_s[wave][mb + r] = rmax[r];
    }
    // ---- pass B: exp(x - rowmax) in LDS, accumulate row sums (2 lanes/row) --
    {
      int row = lane >> 1;
      float rm = rowmax_s[wave][row];
      unsigned short* Pr = &P[wave][row][(lane & 1) * 192];
      float sum = 0.f;
      for (int c = 0; c < 192; c += 8) {
        uint4 pk = *(uint4*)(Pr + c);
        unsigned short* us = (unsigned short*)&pk;
#pragma unroll
        for (int j = 0; j < 8; ++j) {
          float e = __expf(bf2f(us[j]) - rm);
          sum += e;
          us[j] = f2bf(e);
        }
        *(uint4*)(Pr + c) = pk;
      }
      sum += __shfl_xor(sum, 1);
      if ((lane & 1) == 0) rowsum_s[wave][row] = sum;
    }
    // ---- PV: out tile = P (16x384) x V (384x32); normalize in epilogue ----
#pragma unroll
    for (int nt2 = 0; nt2 < 2; ++nt2) {
      v8f ac = {};
#pragma unroll
      for (int kc = 0; kc < 12; ++kc) {
        v16bf pf = load_fragA(&P[wave][0][kc * 32], S_, lane);
        v16bf vf = load_fragB(Vs + (size_t)(nt2 * 16) * S_ + kc * 32, S_, lane);
        ac = wmma_bf16(pf, vf, ac);
      }
#pragma unroll
      for (int r = 0; r < 8; ++r) {
        float rs = rowsum_s[wave][mb + r];
        O[(size_t)(m0 + mb + r) * DIM_ + nt2 * 16 + ncol] = ac[r] / rs;
      }
    }
  }
}

// ---------------- kernel 5: camera mean + LN + MLP -> logits ----------------
__global__ void pool_logits_kernel(const float* __restrict__ a,
                                   const float* __restrict__ g,
                                   const float* __restrict__ bta,
                                   const float* __restrict__ W1,
                                   const float* __restrict__ b1,
                                   const float* __restrict__ W2,
                                   const float* __restrict__ b2,
                                   float* __restrict__ logits) {
  int bl  = blockIdx.x / CAM_;
  int cam = blockIdx.x % CAM_;
  int tid = threadIdx.x;                 // 256
  const float* base = a + ((size_t)bl * S_ + cam * 64) * DIM_;
  __shared__ float am[512];
  __shared__ float red[256];
  float s0 = 0.f, s1 = 0.f;
  for (int p = 0; p < 64; ++p) {
    s0 += base[(size_t)p * DIM_ + tid];
    s1 += base[(size_t)p * DIM_ + tid + 256];
  }
  s0 *= (1.f / 64.f); s1 *= (1.f / 64.f);
  red[tid] = s0 + s1;  __syncthreads();
  for (int o = 128; o; o >>= 1) { if (tid < o) red[tid] += red[tid + o]; __syncthreads(); }
  float mu = red[0] * (1.f / 512.f);
  __syncthreads();
  float d0 = s0 - mu, d1 = s1 - mu;
  red[tid] = d0 * d0 + d1 * d1;  __syncthreads();
  for (int o = 128; o; o >>= 1) { if (tid < o) red[tid] += red[tid + o]; __syncthreads(); }
  float rstd = rsqrtf(red[0] * (1.f / 512.f) + 1e-5f);
  __syncthreads();
  am[tid]       = d0 * rstd * g[tid]       + bta[tid];
  am[tid + 256] = d1 * rstd * g[tid + 256] + bta[tid + 256];
  __syncthreads();
  float acc = b1[tid];
  for (int d = 0; d < 512; ++d) acc += am[d] * W1[(size_t)d * 256 + tid];
  float hv = 0.5f * acc * (1.f + erff(acc * 0.70710678118f));   // exact gelu
  red[tid] = hv * W2[tid];  __syncthreads();
  for (int o = 128; o; o >>= 1) { if (tid < o) red[tid] += red[tid + o]; __syncthreads(); }
  if (tid == 0) logits[blockIdx.x] = red[0] + b2[0];
}

// ---------------- kernel 6: camera softmax + weighted sum -> aw bf16 --------
__global__ void pool_weighted_kernel(const float* __restrict__ a,
                                     const float* __restrict__ logits,
                                     unsigned short* __restrict__ aw) {
  int bl = blockIdx.x;                   // b*L = 32
  int tid = threadIdx.x;
  __shared__ float cw[CAM_];
  if (tid == 0) {
    float lg[CAM_], mx = -3e38f, sm = 0.f;
    for (int n = 0; n < CAM_; ++n) { lg[n] = logits[bl * CAM_ + n]; mx = fmaxf(mx, lg[n]); }
    for (int n = 0; n < CAM_; ++n) { lg[n] = __expf(lg[n] - mx); sm += lg[n]; }
    for (int n = 0; n < CAM_; ++n) cw[n] = lg[n] / sm;
  }
  __syncthreads();
  float c0 = cw[0], c1 = cw[1], c2 = cw[2], c3 = cw[3], c4 = cw[4], c5 = cw[5];
  const float* base = a + (size_t)bl * S_ * DIM_;
  unsigned short* ob = aw + (size_t)bl * 64 * DIM_;
  for (int idx = tid; idx < 64 * DIM_; idx += 256) {
    int p = idx >> 9, d = idx & 511;
    float s = c0 * base[(size_t)(p)       * DIM_ + d]
            + c1 * base[(size_t)(64  + p) * DIM_ + d]
            + c2 * base[(size_t)(128 + p) * DIM_ + d]
            + c3 * base[(size_t)(192 + p) * DIM_ + d]
            + c4 * base[(size_t)(256 + p) * DIM_ + d]
            + c5 * base[(size_t)(320 + p) * DIM_ + d];
    ob[idx] = f2bf(s);
  }
}

// ---------------- workspace layout (~78 MB; a/aw alias dead LN buffers) -----
static constexpr size_t SZ_LN = (size_t)ROWS_ * DIM_ * 2;   // 12,582,912
static constexpr size_t SZ_W  = (size_t)DIM_ * DIM_ * 2;    //    524,288
static constexpr size_t OFF_LNQ = 0;
static constexpr size_t OFF_LNK = OFF_LNQ + SZ_LN;
static constexpr size_t OFF_LNV = OFF_LNK + SZ_LN;
static constexpr size_t OFF_WQT = OFF_LNV + SZ_LN;
static constexpr size_t OFF_WKT = OFF_WQT + SZ_W;
static constexpr size_t OFF_WVT = OFF_WKT + SZ_W;
static constexpr size_t OFF_WPT = OFF_WVT + SZ_W;
static constexpr size_t OFF_QH  = OFF_WPT + SZ_W;
static constexpr size_t OFF_KH  = OFF_QH + SZ_LN;
static constexpr size_t OFF_VT  = OFF_KH + SZ_LN;
static constexpr size_t OFF_LOG = OFF_VT + SZ_LN;
// a (fp32, ROWS_*512*4 = 2*SZ_LN bytes) aliases [OFF_LNQ, OFF_LNV)
// aw (bf16, 2048*512*2 bytes)           aliases OFF_LNV

extern "C" void kernel_launch(void* const* d_in, const int* in_sizes, int n_in,
                              void* d_out, int out_size, void* d_ws, size_t ws_size,
                              hipStream_t stream) {
  (void)in_sizes; (void)n_in; (void)out_size; (void)ws_size;
  const float* q    = (const float*)d_in[0];
  const float* k    = (const float*)d_in[1];
  const float* v    = (const float*)d_in[2];
  const float* q_g  = (const float*)d_in[3];
  const float* q_b  = (const float*)d_in[4];
  const float* Wq   = (const float*)d_in[5];
  const float* bq   = (const float*)d_in[6];
  const float* k_g  = (const float*)d_in[7];
  const float* k_b  = (const float*)d_in[8];
  const float* Wk   = (const float*)d_in[9];
  const float* bk   = (const float*)d_in[10];
  const float* v_g  = (const float*)d_in[11];
  const float* v_b  = (const float*)d_in[12];
  const float* Wv   = (const float*)d_in[13];
  const float* bv   = (const float*)d_in[14];
  const float* lsc  = (const float*)d_in[15];
  const float* Wp   = (const float*)d_in[16];
  const float* bp   = (const float*)d_in[17];
  const float* cs_g = (const float*)d_in[18];
  const float* cs_b = (const float*)d_in[19];
  const float* W1   = (const float*)d_in[20];
  const float* b1   = (const float*)d_in[21];
  const float* W2   = (const float*)d_in[22];
  const float* b2   = (const float*)d_in[23];

  char* ws = (char*)d_ws;
  unsigned short* ln_q = (unsigned short*)(ws + OFF_LNQ);
  unsigned short* ln_k = (unsigned short*)(ws + OFF_LNK);
  unsigned short* ln_v = (unsigned short*)(ws + OFF_LNV);
  unsigned short* wq_t = (unsigned short*)(ws + OFF_WQT);
  unsigned short* wk_t = (unsigned short*)(ws + OFF_WKT);
  unsigned short* wv_t = (unsigned short*)(ws + OFF_WVT);
  unsigned short* wp_t = (unsigned short*)(ws + OFF_WPT);
  unsigned short* qh   = (unsigned short*)(ws + OFF_QH);
  unsigned short* kh   = (unsigned short*)(ws + OFF_KH);
  unsigned short* vt   = (unsigned short*)(ws + OFF_VT);
  float*          lg   = (float*)(ws + OFF_LOG);
  float*          a    = (float*)(ws + OFF_LNQ);   // alias (LN q/k dead by then)
  unsigned short* aw   = (unsigned short*)(ws + OFF_LNV); // alias (LN v dead)

  // 1) weights -> bf16 transposed
  wt_convert_kernel<<<1024, 256, 0, stream>>>(Wq, wq_t);
  wt_convert_kernel<<<1024, 256, 0, stream>>>(Wk, wk_t);
  wt_convert_kernel<<<1024, 256, 0, stream>>>(Wv, wv_t);
  wt_convert_kernel<<<1024, 256, 0, stream>>>(Wp, wp_t);
  // 2) gather + LN + bf16
  ln_gather_kernel<<<ROWS_, 256, 0, stream>>>(q, q_g, q_b, ln_q);
  ln_gather_kernel<<<ROWS_, 256, 0, stream>>>(k, k_g, k_b, ln_k);
  ln_gather_kernel<<<ROWS_, 256, 0, stream>>>(v, v_g, v_b, ln_v);
  // 3) QKV projections (WMMA, 16x128 wave tiles)
  gemm_bf16_kernel<<<dim3(ROWS_/64, 4), 128, 0, stream>>>(ln_q, wq_t, bq, qh, 1);
  gemm_bf16_kernel<<<dim3(ROWS_/64, 4), 128, 0, stream>>>(ln_k, wk_t, bk, kh, 1);
  gemm_bf16_kernel<<<dim3(ROWS_/64, 4), 128, 0, stream>>>(ln_v, wv_t, bv, vt, 2);
  // 4) fused attention (WMMA, K/V staged in LDS via async copy)
  attn_kernel<<<B_*H_*L_, 128, 0, stream>>>(qh, kh, vt, lsc, a);
  // 5) camera pooling MLP -> logits
  pool_logits_kernel<<<B_*L_*CAM_, 256, 0, stream>>>(a, cs_g, cs_b, W1, b1, W2, b2, lg);
  // 6) camera softmax + weighted sum -> aw (bf16)
  pool_weighted_kernel<<<B_*L_, 256, 0, stream>>>(a, lg, aw);
  // 7) final projection (WMMA) -> d_out fp32
  gemm_bf16_kernel<<<dim3((B_*L_*64)/64, 4), 128, 0, stream>>>(aw, wp_t, bp, d_out, 0);
}
